// ContinuousCRF_24610162606599
// MI455X (gfx1250) — matrix operand order repaired and checked by reading, not compiled, not run on hardware
//
#include <hip/hip_runtime.h>
#include <hip/hip_bf16.h>

typedef __attribute__((ext_vector_type(16))) _Float16 v16h;
typedef __attribute__((ext_vector_type(8)))  float    v8f;

#define IMG_H 96
#define IMG_W 96
#define NPIX  (IMG_H * IMG_W)      // 9216
#define NCH   3
#define NPLANES (NCH + 1)          // 3 real channels + 1 zero plane for lanes n>=3
#define ROWS_PER_BLOCK 8           // image rows per workgroup
#define HALO 5                     // window radius (dist <= 5)
#define LROWS (ROWS_PER_BLOCK + 2*HALO)   // 18 LDS rows
#define LSTRIDE 112                // 96 + 2*HALO + slack for K=26..31 (weight 0)
// 112*2B = 224B row stride: multiple of 32B -> aligned v16h loads
// plane size 18*112 = 2016 elems = 4032B, multiple of 32B -> aligned

// ---------------------------------------------------------------------------
// q0 = softmax(unary) over the 3 channels, stored as f16
// ---------------------------------------------------------------------------
__global__ void crf_init_kernel(const float* __restrict__ unary,
                                _Float16* __restrict__ q) {
    int p = blockIdx.x * blockDim.x + threadIdx.x;
    if (p >= NPIX) return;
    float l0 = unary[p];
    float l1 = unary[NPIX + p];
    float l2 = unary[2 * NPIX + p];
    float mx = fmaxf(l0, fmaxf(l1, l2));
    float e0 = __expf(l0 - mx), e1 = __expf(l1 - mx), e2 = __expf(l2 - mx);
    float inv = 1.0f / (e0 + e1 + e2);
    q[p]            = (_Float16)(e0 * inv);
    q[NPIX + p]     = (_Float16)(e1 * inv);
    q[2 * NPIX + p] = (_Float16)(e2 * inv);
}

// ---------------------------------------------------------------------------
// One mean-field iteration:
//   messages = pairwise-stencil(q)   [11x V_WMMA_F32_16X16X32_F16 per 16-px tile]
//   q' = softmax(unary + compat @ messages)
// Each block handles ROWS_PER_BLOCK image rows; q tile (+halo) staged in LDS.
// Lanes whose D-column n >= 3 read B from a zero-filled LDS plane so the
// inner loop has no EXEC divergence and no zero-fill VALU.
// ---------------------------------------------------------------------------
__global__ void __launch_bounds__(256)
crf_iter_kernel(const float* __restrict__ unary,
                const float* __restrict__ compat,
                const _Float16* __restrict__ qin,
                _Float16* __restrict__ qout,
                float* __restrict__ outf32) {
    __shared__ alignas(32) _Float16 sq[NPLANES][LROWS][LSTRIDE];  // 16128 B
    __shared__ float smsg[NCH][ROWS_PER_BLOCK][IMG_W];            //  9216 B

    const int y0   = blockIdx.x * ROWS_PER_BLOCK;
    const int lane = threadIdx.x & 31;
    const int wave = threadIdx.x >> 5;
    const int m    = lane & 15;          // A-row / B-column / D-column index
    const bool hi  = lane >= 16;
    const int kbase = hi ? 16 : 0;       // B K-range for this half-wave
    const int cch  = (m < NCH) ? m : NCH; // channel plane (3 = zero plane)

    // --- Build the 11 constant Toeplitz A-matrices (weights) in registers ---
    // 16-bit A 16x32 layout: lane m = row; element 2j+t -> K = 16*half + 2j + t (contiguous, same as B).
    v16h aMat[11];
#pragma unroll
    for (int dy = 0; dy < 11; ++dy) {
        v16h a;
#pragma unroll
        for (int j = 0; j < 8; ++j) {
            int K = 2 * j + (hi ? 16 : 0);
#pragma unroll
            for (int t = 0; t < 2; ++t) {
                int dx  = (K + t) - HALO - m;     // input col - output col
                int ddy = dy - HALO;
                int r2  = ddy * ddy + dx * dx;
                float wv = (r2 > 0 && r2 <= HALO * HALO)
                               ? __expf(-__fsqrt_rn((float)r2))   // THETA_BETA=1
                               : 0.0f;
                a[2 * j + t] = (_Float16)wv;
            }
        }
        aMat[dy] = a;
    }

    // --- Stage q tile + zero halo into LDS; plane 3 is all zeros ---
    for (int e = threadIdx.x; e < NPLANES * LROWS * LSTRIDE; e += blockDim.x) {
        int c   = e / (LROWS * LSTRIDE);
        int rem = e % (LROWS * LSTRIDE);
        int r   = rem / LSTRIDE;
        int col = rem % LSTRIDE;
        int gy = y0 - HALO + r;
        int gx = col - HALO;
        _Float16 v = (_Float16)0.0f;
        if (c < NCH && gy >= 0 && gy < IMG_H && gx >= 0 && gx < IMG_W)
            v = qin[c * NPIX + gy * IMG_W + gx];
        sq[c][r][col] = v;
    }
    __syncthreads();

    // --- Stencil as WMMA: 48 tiles (8 rows x 6 segments of 16 px), 8 waves ---
    for (int t = wave; t < ROWS_PER_BLOCK * 6; t += 8) {
        int ty = t / 6;
        int x0 = (t % 6) * 16;
        // per-lane B base: channel plane (or zero plane), LDS row ty, col x0+kbase
        const _Float16* bp = &sq[cch][ty][x0 + kbase];
        v8f acc = {};
#pragma unroll
        for (int dy = 0; dy < 11; ++dy) {
            v16h b = *(const v16h*)(bp + dy * LSTRIDE);  // ds_load offset dy*224
            acc = __builtin_amdgcn_wmma_f32_16x16x32_f16(
                /*neg_a=*/false, aMat[dy], /*neg_b=*/false, b,
                /*c_mod=*/(short)0, acc, /*reuse_a=*/false, /*reuse_b=*/false);
        }
        // D[mrow, n] = messages[channel n, pixel x0+mrow]; lane holds column n=m
        if (m < NCH) {
            int mb = hi ? 8 : 0;
#pragma unroll
            for (int j = 0; j < 8; ++j)
                smsg[m][ty][x0 + j + mb] = acc[j];
        }
    }
    __syncthreads();

    // --- compat mix + softmax per pixel, write q' (f16) and optional f32 out ---
    float cm[9];
#pragma unroll
    for (int i = 0; i < 9; ++i) cm[i] = compat[i];

    for (int p = threadIdx.x; p < ROWS_PER_BLOCK * IMG_W; p += blockDim.x) {
        int ry = p / IMG_W;
        int x  = p % IMG_W;
        int gp = (y0 + ry) * IMG_W + x;
        float m0 = smsg[0][ry][x];
        float m1 = smsg[1][ry][x];
        float m2 = smsg[2][ry][x];
        // logits = (unary + compat @ messages) / TEMP, TEMP = 1
        float l0 = unary[gp]            + cm[0]*m0 + cm[1]*m1 + cm[2]*m2;
        float l1 = unary[NPIX + gp]     + cm[3]*m0 + cm[4]*m1 + cm[5]*m2;
        float l2 = unary[2*NPIX + gp]   + cm[6]*m0 + cm[7]*m1 + cm[8]*m2;
        float mx = fmaxf(l0, fmaxf(l1, l2));
        float e0 = __expf(l0 - mx), e1 = __expf(l1 - mx), e2 = __expf(l2 - mx);
        float inv = 1.0f / (e0 + e1 + e2);
        float q0 = e0 * inv, q1 = e1 * inv, q2 = e2 * inv;
        qout[gp]            = (_Float16)q0;
        qout[NPIX + gp]     = (_Float16)q1;
        qout[2 * NPIX + gp] = (_Float16)q2;
        if (outf32) {
            outf32[gp]            = q0;
            outf32[NPIX + gp]     = q1;
            outf32[2 * NPIX + gp] = q2;
        }
    }
}

extern "C" void kernel_launch(void* const* d_in, const int* in_sizes, int n_in,
                              void* d_out, int out_size, void* d_ws, size_t ws_size,
                              hipStream_t stream) {
    const float* unary  = (const float*)d_in[0];   // [3][96][96] f32
    const float* compat = (const float*)d_in[1];   // [3][3] f32
    float* out = (float*)d_out;                    // [3][96][96] f32

    _Float16* qA = (_Float16*)d_ws;                // 3*9216 f16
    _Float16* qB = qA + NCH * NPIX;                // 3*9216 f16

    const int ITER_BLOCKS = IMG_H / ROWS_PER_BLOCK;  // 12

    crf_init_kernel<<<(NPIX + 255) / 256, 256, 0, stream>>>(unary, qA);
    // NUM_ITERS = 5, ping-pong q through scratch; last one also writes f32 out
    crf_iter_kernel<<<ITER_BLOCKS, 256, 0, stream>>>(unary, compat, qA, qB, nullptr);
    crf_iter_kernel<<<ITER_BLOCKS, 256, 0, stream>>>(unary, compat, qB, qA, nullptr);
    crf_iter_kernel<<<ITER_BLOCKS, 256, 0, stream>>>(unary, compat, qA, qB, nullptr);
    crf_iter_kernel<<<ITER_BLOCKS, 256, 0, stream>>>(unary, compat, qB, qA, nullptr);
    crf_iter_kernel<<<ITER_BLOCKS, 256, 0, stream>>>(unary, compat, qA, qB, out);
}
